// SimCLR_15006615733295
// MI455X (gfx1250) — matrix operand order repaired and checked by reading, not compiled, not run on hardware
//
#include <hip/hip_runtime.h>
#include <stdint.h>

// ---------------------------------------------------------------------------
// Types for CDNA5 WMMA
// ---------------------------------------------------------------------------
typedef __attribute__((ext_vector_type(16))) __bf16 v16bf;
typedef __attribute__((ext_vector_type(8)))  float  v8f;

union FragBF { uint32_t u[8]; v16bf v; };

// fp32 -> bf16 (round to nearest even), packed pair helpers
__device__ __forceinline__ uint32_t bf16b(float f) {
    uint32_t u = __float_as_uint(f);
    return (u + 0x7FFFu + ((u >> 16) & 1u)) >> 16;
}
__device__ __forceinline__ uint32_t pkbf2(float a, float b) {
    return bf16b(a) | (bf16b(b) << 16);
}

// CDNA5 async global->LDS copy (16B per lane), tracked by ASYNCcnt.
// GVS addressing: mem = SGPR_base + VGPR_offset; LDS dest addr from VGPR.
__device__ __forceinline__ void async_cp16(uint32_t lds_addr, uint32_t gbl_off,
                                           const void* base) {
    asm volatile("global_load_async_to_lds_b128 %0, %1, %2"
                 :: "v"(lds_addr), "v"(gbl_off), "s"(base)
                 : "memory");
}
__device__ __forceinline__ void wait_async0() {
    asm volatile("s_wait_asynccnt 0" ::: "memory");
}
__device__ __forceinline__ uint32_t lds_off32(const void* p) {
    return (uint32_t)(uintptr_t)p;  // LDS aperture lives in addr[63:32]
}

// ---------------------------------------------------------------------------
// Tiny utility kernels
// ---------------------------------------------------------------------------
__global__ void zero_kernel(float* p, int n) {
    int i = blockIdx.x * blockDim.x + threadIdx.x;
    if (i < n) p[i] = 0.0f;
}

// fp32 -> bf16-pair-packed u32 stream (pairs contiguous along row)
__global__ void pack_x_kernel(const float* __restrict__ X, uint32_t* __restrict__ Xp,
                              int total4) {
    for (int i = blockIdx.x * blockDim.x + threadIdx.x; i < total4;
         i += gridDim.x * blockDim.x) {
        const float4 f = ((const float4*)X)[i];
        Xp[2 * i]     = pkbf2(f.x, f.y);
        Xp[2 * i + 1] = pkbf2(f.z, f.w);
    }
}

// W (K x N fp32, row major) -> Wp ((K/2) x N u32, each u32 = bf16 pair (k, k+1))
__global__ void pack_w_kernel(const float* __restrict__ W, uint32_t* __restrict__ Wp,
                              int K, int N) {
    int total = (K >> 1) * N;
    for (int idx = blockIdx.x * blockDim.x + threadIdx.x; idx < total;
         idx += gridDim.x * blockDim.x) {
        int kp = idx / N;
        int n  = idx - kp * N;
        Wp[idx] = pkbf2(W[(2 * kp) * N + n], W[(2 * kp + 1) * N + n]);
    }
}

// BatchNorm (training-mode biased stats, per input half) + ReLU, emit bf16
__global__ void bn_relu_kernel(const float* __restrict__ Y, const float* __restrict__ stats,
                               const float* __restrict__ g, const float* __restrict__ beta,
                               uint16_t* __restrict__ H, int M, int N) {
    const int    halfM = M >> 1;
    const float  invB  = 1.0f / (float)halfM;
    const size_t total = (size_t)M * N;
    for (size_t idx = (size_t)blockIdx.x * blockDim.x + threadIdx.x; idx < total;
         idx += (size_t)gridDim.x * blockDim.x) {
        int n   = (int)(idx % N);
        int row = (int)(idx / N);
        const float* sb = stats + ((row >= halfM) ? 2 * N : 0);
        float mu  = sb[n] * invB;
        float var = sb[N + n] * invB - mu * mu;
        float h = (Y[idx] - mu) * rsqrtf(var + 1e-5f) * g[n] + beta[n];
        h = fmaxf(h, 0.0f);
        H[idx] = (uint16_t)bf16b(h);
    }
}

// ---------------------------------------------------------------------------
// BF16 WMMA GEMM with async double-buffered LDS staging.
//   C[M x N] = A[M x K] @ B[K x N] + bias; A/B bf16 pair-packed u32 in global.
//   EPI=0 (STATS): write fp32 C, accumulate per-column sum & sum-of-squares
//                  into aux (split by input half: [sel][sum|sq][N])
//   EPI=1 (NORM) : N == 128 == tile width; add bias, row-L2-normalize,
//                  write Z, accumulate global column sum S into aux
// Tile: BM=128, BN=128, KC=64; 256 threads = 8 wave32 (2 M x 4 N),
// each wave computes a 64x32 sub-tile = 4x2 fragments of v_wmma_f32_16x16x32_bf16.
// ---------------------------------------------------------------------------
#define BM 128
#define BN 128
#define KC 64
#define AS_U32 37   // padded u32 row stride for A tile (128 rows x 32 pairs)
#define BS_U32 132  // padded u32 row stride for B tile (32 pair-rows x 128 cols)

template <int EPI>
__global__ __launch_bounds__(256) void gemm_kernel(
    const uint32_t* __restrict__ A0, const uint32_t* __restrict__ A1,
    const uint32_t* __restrict__ Bpk, const float* __restrict__ bias,
    float* __restrict__ out, float* __restrict__ aux,
    int M, int N, int K) {
    __shared__ uint32_t As[2][BM * AS_U32];
    __shared__ uint32_t Bs[2][(KC / 2) * BS_U32];
    __shared__ float    redA[BM];
    __shared__ float    redB[BN];

    const int tid  = threadIdx.x;
    const int lane = tid & 31;
    const int w    = tid >> 5;
    const int wm   = w & 1;   // 0..1 : 64-row slab
    const int wn   = w >> 1;  // 0..3 : 32-col slab
    const int hl   = lane >> 4;
    const int l15  = lane & 15;

    const int r0    = blockIdx.y * BM;
    const int c0    = blockIdx.x * BN;
    const int halfM = M >> 1;
    const int sel   = (r0 >= halfM) ? 1 : 0;
    const int rloc  = r0 - sel * halfM;

    const uint32_t* Ab  = sel ? A1 : A0;
    const int       ldu = K >> 1;  // u32 per A row

    // per-thread staging coordinates (16B = 4 u32 per lane, 4 issues/tile)
    // A: 128 rows x 32 u32 ; B: 32 pair-rows x 128 u32
    auto stage = [&](int buf, int k0) {
        const uint32_t asb = lds_off32(&As[buf][0]);
        const uint32_t bsb = lds_off32(&Bs[buf][0]);
#pragma unroll
        for (int j = 0; j < 4; ++j) {
            int lid = tid + (j << 8);
            int ar  = lid >> 3;          // A row 0..127
            int ap  = (lid & 7) << 2;    // A u32 col 0,4,..,28
            async_cp16(asb + (uint32_t)(ar * AS_U32 + ap) * 4u,
                       (uint32_t)(((rloc + ar) * ldu + (k0 >> 1) + ap) * 4),
                       Ab);
            int bkp = lid >> 5;          // B pair-row 0..31
            int bn  = (lid & 31) << 2;   // B u32 col 0,4,..,124
            async_cp16(bsb + (uint32_t)(bkp * BS_U32 + bn) * 4u,
                       (uint32_t)(((((k0 >> 1) + bkp) * N) + c0 + bn) * 4),
                       Bpk);
        }
    };

    v8f acc[4][2];
    {
        v8f zz = {0.f, 0.f, 0.f, 0.f, 0.f, 0.f, 0.f, 0.f};
        for (int fm = 0; fm < 4; ++fm)
            for (int fn = 0; fn < 2; ++fn) acc[fm][fn] = zz;
    }

    const int nk = K / KC;
    stage(0, 0);
    wait_async0();
    __syncthreads();

    for (int kb = 0; kb < nk; ++kb) {
        const int cur = kb & 1;
        if (kb + 1 < nk) stage(cur ^ 1, (kb + 1) * KC);  // overlap with WMMA

        // ---- 2 WMMA k-steps (K=32 each) on buffer `cur` ----
#pragma unroll
        for (int ks = 0; ks < 2; ++ks) {
            const int kk2 = ks * 16;  // base k/2 within chunk
            FragBF b[2];
#pragma unroll
            for (int fn = 0; fn < 2; ++fn) {
                int n = wn * 32 + fn * 16 + l15;
#pragma unroll
                for (int v = 0; v < 8; ++v)
                    b[fn].u[v] = Bs[cur][(kk2 + hl * 8 + v) * BS_U32 + n];
            }
            FragBF a[4];
#pragma unroll
            for (int fm = 0; fm < 4; ++fm) {
                int m = wm * 64 + fm * 16 + l15;
#pragma unroll
                for (int v = 0; v < 8; ++v) {
                    int k2 = kk2 + hl * 8 + v;
                    a[fm].u[v] = As[cur][m * AS_U32 + k2];
                }
            }
#pragma unroll
            for (int fm = 0; fm < 4; ++fm)
#pragma unroll
                for (int fn = 0; fn < 2; ++fn)
                    acc[fm][fn] = __builtin_amdgcn_wmma_f32_16x16x32_bf16(
                        false, a[fm].v, false, b[fn].v, (short)0, acc[fm][fn],
                        false, false);
        }
        wait_async0();     // my async copies into the other buffer landed
        __syncthreads();   // everyone done reading `cur` / writing `cur^1`
    }

    // ---- epilogue ----
    if (EPI == 0) {
        if (tid < BN) { redA[tid] = 0.f; redB[tid] = 0.f; }
        __syncthreads();
        for (int fn = 0; fn < 2; ++fn) {
            int   nl   = wn * 32 + fn * 16 + l15;
            float bcol = bias[c0 + nl];
            float s = 0.f, s2 = 0.f;
            for (int fm = 0; fm < 4; ++fm) {
                int mbase = wm * 64 + fm * 16 + hl * 8;
                for (int v = 0; v < 8; ++v) {
                    float y = acc[fm][fn][v] + bcol;
                    out[(size_t)(r0 + mbase + v) * N + c0 + nl] = y;
                    s += y;
                    s2 += y * y;
                }
            }
            atomicAdd(&redA[nl], s);
            atomicAdd(&redB[nl], s2);
        }
        __syncthreads();
        if (tid < BN) {
            float* sbase = aux + sel * 2 * N;
            atomicAdd(&sbase[c0 + tid], redA[tid]);
            atomicAdd(&sbase[N + c0 + tid], redB[tid]);
        }
    } else {
        // NORM: full row (N == BN == 128) present in this block
        if (tid < BM) { redA[tid] = 0.f; redB[tid] = 0.f; }
        __syncthreads();
        for (int fn = 0; fn < 2; ++fn) {
            float bcol = bias[c0 + wn * 32 + fn * 16 + l15];
            for (int fm = 0; fm < 4; ++fm)
                for (int v = 0; v < 8; ++v) acc[fm][fn][v] += bcol;
        }
        // row sum-of-squares: reduce across the 16 N-lanes, then LDS across waves
        for (int fm = 0; fm < 4; ++fm) {
            float ss[8];
            for (int v = 0; v < 8; ++v)
                ss[v] = acc[fm][0][v] * acc[fm][0][v] + acc[fm][1][v] * acc[fm][1][v];
            for (int msk = 1; msk < 16; msk <<= 1)
                for (int v = 0; v < 8; ++v) ss[v] += __shfl_xor(ss[v], msk, 32);
            if (l15 == 0) {
                int mbase = wm * 64 + fm * 16 + hl * 8;
                for (int v = 0; v < 8; ++v) atomicAdd(&redA[mbase + v], ss[v]);
            }
        }
        __syncthreads();
        // normalize rows, write Z, accumulate column sums for S
        for (int fn = 0; fn < 2; ++fn) {
            int   nl = wn * 32 + fn * 16 + l15;
            float cs = 0.f;
            for (int fm = 0; fm < 4; ++fm) {
                int mbase = wm * 64 + fm * 16 + hl * 8;
                for (int v = 0; v < 8; ++v) {
                    float nrm = sqrtf(redA[mbase + v]);
                    float zv  = acc[fm][fn][v] / fmaxf(nrm, 1e-12f);
                    out[(size_t)(r0 + mbase + v) * N + c0 + nl] = zv;
                    cs += zv;
                }
            }
            atomicAdd(&redB[nl], cs);
        }
        __syncthreads();
        if (tid < BN) atomicAdd(&aux[c0 + tid], redB[tid]);
    }
}

// ---------------------------------------------------------------------------
// Loss: per row i, pos = z1.z2 ; rowsum_k = z_k . S ; denom = (rowsum-1)/T
// loss_i + loss_{i+B} = -2*pos/T + log(d1) + log(d2); mean over 2B via atomic.
// ---------------------------------------------------------------------------
__global__ __launch_bounds__(128) void loss_kernel(const float* __restrict__ Z,
                                                   const float* __restrict__ S,
                                                   float* __restrict__ out, int Bv, int D) {
    __shared__ float q0[128], q1[128], q2[128];
    const int i = blockIdx.x;
    const int t = threadIdx.x;
    float z1 = Z[(size_t)i * D + t];
    float z2 = Z[(size_t)(i + Bv) * D + t];
    float s  = S[t];
    q0[t] = z1 * z2;
    q1[t] = z1 * s;
    q2[t] = z2 * s;
    __syncthreads();
    for (int off = 64; off > 0; off >>= 1) {
        if (t < off) {
            q0[t] += q0[t + off];
            q1[t] += q1[t + off];
            q2[t] += q2[t + off];
        }
        __syncthreads();
    }
    if (t == 0) {
        const float invT = 1.0f / 0.07f;
        float c = -2.0f * q0[0] * invT
                  + logf(fmaxf((q1[0] - 1.0f) * invT, 1e-30f))
                  + logf(fmaxf((q2[0] - 1.0f) * invT, 1e-30f));
        atomicAdd(out, c / (2.0f * (float)Bv));
    }
}

// ---------------------------------------------------------------------------
// Launch
// ---------------------------------------------------------------------------
extern "C" void kernel_launch(void* const* d_in, const int* in_sizes, int n_in,
                              void* d_out, int out_size, void* d_ws, size_t ws_size,
                              hipStream_t stream) {
    (void)in_sizes; (void)n_in; (void)out_size; (void)ws_size;
    const float* x1  = (const float*)d_in[0];
    const float* x2  = (const float*)d_in[1];
    const float* W1  = (const float*)d_in[2];
    const float* b1  = (const float*)d_in[3];
    const float* g1  = (const float*)d_in[4];
    const float* be1 = (const float*)d_in[5];
    const float* W2  = (const float*)d_in[6];
    const float* b2  = (const float*)d_in[7];
    const float* g2  = (const float*)d_in[8];
    const float* be2 = (const float*)d_in[9];
    const float* W3  = (const float*)d_in[10];
    const float* b3  = (const float*)d_in[11];
    float*       out = (float*)d_out;

    const int Mtot = 16384, B = 8192;
    char*  ws  = (char*)d_ws;
    size_t off = 0;
    auto alloc = [&](size_t bytes) -> void* {
        void* p = ws + off;
        off = (off + bytes + 255) & ~(size_t)255;
        return p;
    };
    float*    Y   = (float*)alloc((size_t)Mtot * 256 * 4);
    uint16_t* H   = (uint16_t*)alloc((size_t)Mtot * 256 * 2);
    float*    Z   = (float*)alloc((size_t)Mtot * 128 * 4);
    uint32_t* Xbf = (uint32_t*)alloc((size_t)Mtot * 512 * 2);  // bf16 x1|x2
    uint32_t* W1p = (uint32_t*)alloc((size_t)256 * 256 * 4);
    uint32_t* W2p = (uint32_t*)alloc((size_t)128 * 256 * 4);
    uint32_t* W3p = (uint32_t*)alloc((size_t)128 * 128 * 4);
    float*    st  = (float*)alloc((size_t)(1024 + 1024 + 128) * 4);
    float*    stats1 = st;
    float*    stats2 = st + 1024;
    float*    S      = st + 2048;

    zero_kernel<<<(2176 + 255) / 256, 256, 0, stream>>>(st, 2176);
    zero_kernel<<<1, 32, 0, stream>>>(out, 1);

    const int x4 = B * 512 / 4;  // float4 count per input
    pack_x_kernel<<<2048, 256, 0, stream>>>(x1, Xbf, x4);
    pack_x_kernel<<<2048, 256, 0, stream>>>(x2, Xbf + (size_t)B * 256, x4);
    pack_w_kernel<<<64, 256, 0, stream>>>(W1, W1p, 512, 256);
    pack_w_kernel<<<32, 256, 0, stream>>>(W2, W2p, 256, 256);
    pack_w_kernel<<<16, 256, 0, stream>>>(W3, W3p, 256, 128);

    // layer 1: (16384 x 512) @ (512 x 256)
    gemm_kernel<0><<<dim3(2, Mtot / BM), 256, 0, stream>>>(
        Xbf, Xbf + (size_t)B * 256, W1p, b1, Y, stats1, Mtot, 256, 512);
    bn_relu_kernel<<<4096, 256, 0, stream>>>(Y, stats1, g1, be1, H, Mtot, 256);

    // layer 2: (16384 x 256) @ (256 x 256)
    gemm_kernel<0><<<dim3(2, Mtot / BM), 256, 0, stream>>>(
        (const uint32_t*)H, (const uint32_t*)H + (size_t)B * 128, W2p, b2, Y,
        stats2, Mtot, 256, 256);
    bn_relu_kernel<<<4096, 256, 0, stream>>>(Y, stats2, g2, be2, H, Mtot, 256);

    // layer 3 + normalize + S accumulation: (16384 x 256) @ (256 x 128)
    gemm_kernel<1><<<dim3(1, Mtot / BM), 256, 0, stream>>>(
        (const uint32_t*)H, (const uint32_t*)H + (size_t)B * 128, W3p, b3, Z, S,
        Mtot, 128, 256);

    // loss
    loss_kernel<<<B, 128, 0, stream>>>(Z, S, out, B, 128);
}
